// BiDAF_13812614824291
// MI455X (gfx1250) — compile-verified, matrix operand order checked by reading, not run on hardware
//
#include <hip/hip_runtime.h>
#include <hip/hip_bf16.h>

// ---------------------------------------------------------------------------
// BiDAF forward for gfx1250 (MI455X, wave32, WMMA).
// Heavy matmuls (LSTM input projections, per-step recurrent gate GEMMs) use
// v_wmma_f32_16x16x32_f16 with f32 accumulation. Recurrent state lives in LDS.
// ---------------------------------------------------------------------------

typedef __attribute__((ext_vector_type(16))) _Float16 v16h;
typedef __attribute__((ext_vector_type(8)))  float    v8f;

#define B_     32
#define CLEN_  400
#define QLEN_  48
#define WLEN_  16
#define WDIM_  300
#define CEMB_  16
#define COUT_  100
#define CK_    3
#define HID_   100
#define G4_    400   // 4*HID (all GEMM N dims == 400 == 5 tiles of 16 * grid 5)
#define EMB_   400   // WDIM + COUT
#define EMBP_  416   // EMB padded to multiple of 32 (WMMA K step)
#define HP_    128   // HID padded
#define D2_    200   // 2*HID
#define D2P_   224   // D2 padded
#define D8_    800   // 8*HID

__device__ __forceinline__ float sigmoidf_(float x) { return 1.0f / (1.0f + expf(-x)); }

// Build a 16x32 f16 WMMA fragment from a K-contiguous row slice.
// p must already include the per-lane khalf offset; lowers to 2x b128 loads.
__device__ __forceinline__ v16h load_frag(const _Float16* __restrict__ p)
{
    v16h r;
#pragma unroll
    for (int j = 0; j < 8; ++j) { r[j] = p[j]; r[8 + j] = p[16 + j]; }
    return r;
}

__device__ __forceinline__ v8f zero8()
{
    v8f z;
#pragma unroll
    for (int i = 0; i < 8; ++i) z[i] = 0.0f;
    return z;
}

// ------------------------- WMMA GEMM: C[M,N] = A[M,Kp] * W[N,Kp]^T ---------
// A, W are f16 row-major (K padded to multiple of 32 with zeros), C is f32.
// One wave per block; each wave owns a 16(M) x 80(N) tile (5 accumulators).
// Requires N % 80 == 0 (all GEMMs here have N = 400); grid = (N/80, M/16).
__global__ __launch_bounds__(32)
void gemm_wmma_f16(const _Float16* __restrict__ A, const _Float16* __restrict__ W,
                   float* __restrict__ C, int M, int N, int Kp)
{
    const int lane  = threadIdx.x & 31;
    const int m0    = blockIdx.y * 16;
    const int n0b   = blockIdx.x * 80;
    const int khalf = (lane & 16) ? 8 : 0;   // K sub-block per lane half (A/B frag)
    const int rsel  = (lane & 16) ? 8 : 0;   // M offset per lane half (C/D frag)
    const int lrow  = lane & 15;

    v8f acc0 = zero8(), acc1 = zero8(), acc2 = zero8(), acc3 = zero8(), acc4 = zero8();

    const _Float16* arow = A + (size_t)(m0 + lrow) * Kp + khalf;
    const _Float16* wrow = W + (size_t)(n0b + lrow) * Kp + khalf;
    const size_t wstep = (size_t)16 * Kp;    // next 16-column tile of W

    for (int k0 = 0; k0 < Kp; k0 += 32) {
        // Batch all loads first so they clause + overlap, then run the WMMAs.
        const v16h a  = load_frag(arow + k0);
        const v16h b0 = load_frag(wrow + k0);
        const v16h b1 = load_frag(wrow + wstep + k0);
        const v16h b2 = load_frag(wrow + 2 * wstep + k0);
        const v16h b3 = load_frag(wrow + 3 * wstep + k0);
        const v16h b4 = load_frag(wrow + 4 * wstep + k0);
        acc0 = __builtin_amdgcn_wmma_f32_16x16x32_f16(false, a, false, b0, (short)0, acc0, false, false);
        acc1 = __builtin_amdgcn_wmma_f32_16x16x32_f16(false, a, false, b1, (short)0, acc1, false, false);
        acc2 = __builtin_amdgcn_wmma_f32_16x16x32_f16(false, a, false, b2, (short)0, acc2, false, false);
        acc3 = __builtin_amdgcn_wmma_f32_16x16x32_f16(false, a, false, b3, (short)0, acc3, false, false);
        acc4 = __builtin_amdgcn_wmma_f32_16x16x32_f16(false, a, false, b4, (short)0, acc4, false, false);
    }

    float* crow = C + (size_t)(m0 + rsel) * N + lrow;
#pragma unroll
    for (int i = 0; i < 8; ++i) {
        float* c = crow + (size_t)i * N;
        c[n0b]      = acc0[i];
        c[n0b + 16] = acc1[i];
        c[n0b + 32] = acc2[i];
        c[n0b + 48] = acc3[i];
        c[n0b + 64] = acc4[i];
    }
}

// --------------------- Bidirectional LSTM recurrence -----------------------
// grid = (2 batch-halves, 2 directions). Per block: 16 batch rows, 8 waves.
// Per step: gates[16,400] = h16[16,128] x Whh[400,128]^T via WMMA, then
// elementwise LSTM cell; h/c state in LDS. Pf/Pb are precomputed x*Wih^T.
__global__ __launch_bounds__(256)
void lstm_birec(const float* __restrict__ Pf, const float* __restrict__ Pb,
                const _Float16* __restrict__ WhhF, const _Float16* __restrict__ WhhB,
                const float* __restrict__ biasF, const float* __restrict__ biasB,
                float* __restrict__ Hout, int T)
{
    __shared__ _Float16 h16[16][HP_];
    __shared__ float    gates[16][G4_];
    __shared__ float    cbuf[16][HID_];

    const int tid   = threadIdx.x;
    const int lane  = tid & 31;
    const int wave  = tid >> 5;
    const int khalf = (lane & 16) ? 8 : 0;
    const int rsel  = (lane & 16) ? 8 : 0;
    const int lrow  = lane & 15;
    const int bbase = blockIdx.x * 16;

    const int dir = blockIdx.y;
    const float*    P    = dir ? Pb    : Pf;
    const _Float16* Whh  = dir ? WhhB  : WhhF;
    const float*    bias = dir ? biasB : biasF;
    const int t0     = dir ? (T - 1) : 0;
    const int dt     = dir ? -1 : 1;
    const int coloff = dir ? HID_ : 0;

    for (int i = tid; i < 16 * HP_;  i += 256) ((_Float16*)h16)[i] = (_Float16)0.0f;
    for (int i = tid; i < 16 * HID_; i += 256) ((float*)cbuf)[i] = 0.0f;
    __syncthreads();

    int t = t0;
    for (int s = 0; s < T; ++s, t += dt) {
        // A fragments from recurrent hidden state (LDS)
        v16h a0, a1, a2, a3;
        {
            const _Float16* hb = &h16[lrow][khalf];
            a0 = load_frag(hb);
            a1 = load_frag(hb + 32);
            a2 = load_frag(hb + 64);
            a3 = load_frag(hb + 96);
        }
        // 25 N-tiles of 16 gates spread over 8 waves
        for (int tile = wave; tile < 25; tile += 8) {
            const int n0 = tile * 16;
            const _Float16* wb = Whh + (size_t)(n0 + lrow) * HP_ + khalf;
            const v16h b0 = load_frag(wb);
            const v16h b1 = load_frag(wb + 32);
            const v16h b2 = load_frag(wb + 64);
            const v16h b3 = load_frag(wb + 96);
            v8f acc = zero8();
            acc = __builtin_amdgcn_wmma_f32_16x16x32_f16(false, a0, false, b0, (short)0, acc, false, false);
            acc = __builtin_amdgcn_wmma_f32_16x16x32_f16(false, a1, false, b1, (short)0, acc, false, false);
            acc = __builtin_amdgcn_wmma_f32_16x16x32_f16(false, a2, false, b2, (short)0, acc, false, false);
            acc = __builtin_amdgcn_wmma_f32_16x16x32_f16(false, a3, false, b3, (short)0, acc, false, false);
#pragma unroll
            for (int i = 0; i < 8; ++i)
                gates[rsel + i][n0 + lrow] = acc[i];
        }
        __syncthreads();

        // LSTM cell (f32): c = sig(f)*c + sig(i)*tanh(g); h = sig(o)*tanh(c)
        for (int idx = tid; idx < 16 * HID_; idx += 256) {
            const int m = idx / HID_;
            const int j = idx - m * HID_;
            const size_t row = (size_t)(bbase + m) * T + t;
            const float* p = P + row * G4_;
            const float ig = gates[m][j]            + p[j]            + bias[j];
            const float fg = gates[m][HID_ + j]     + p[HID_ + j]     + bias[HID_ + j];
            const float gg = gates[m][2 * HID_ + j] + p[2 * HID_ + j] + bias[2 * HID_ + j];
            const float og = gates[m][3 * HID_ + j] + p[3 * HID_ + j] + bias[3 * HID_ + j];
            const float c  = sigmoidf_(fg) * cbuf[m][j] + sigmoidf_(ig) * tanhf(gg);
            const float h  = sigmoidf_(og) * tanhf(c);
            cbuf[m][j] = c;
            h16[m][j]  = (_Float16)h;
            Hout[row * D2_ + coloff + j] = h;
        }
        __syncthreads();
    }
}

// ---------------- Word + char-CNN embedding (writes f16, K padded) ---------
__global__ __launch_bounds__(128)
void embed_kernel(const int* __restrict__ wids, const int* __restrict__ cids,
                  const float* __restrict__ wv, const float* __restrict__ ctab,
                  const float* __restrict__ cw, const float* __restrict__ cb,
                  _Float16* __restrict__ emb)
{
    __shared__ float e[WLEN_][CEMB_];
    const int tid = threadIdx.x;
    const size_t row = blockIdx.x;

    for (int idx = tid; idx < WLEN_ * CEMB_; idx += 128) {
        const int ch = idx >> 4, d = idx & 15;
        const int cid = cids[row * WLEN_ + ch];
        e[ch][d] = ctab[cid * CEMB_ + d];
    }
    __syncthreads();

    if (tid < COUT_) {
        float best = 0.0f;  // relu output >= 0
        for (int pp = 0; pp < WLEN_ - CK_ + 1; ++pp) {
            float acc = cb[tid];
            for (int k = 0; k < CK_; ++k)
                for (int i = 0; i < CEMB_; ++i)
                    acc += e[pp + k][i] * cw[tid * (CEMB_ * CK_) + i * CK_ + k];
            best = fmaxf(best, fmaxf(acc, 0.0f));
        }
        emb[row * EMBP_ + WDIM_ + tid] = (_Float16)best;
    }
    const int wid = wids[row];
    for (int d = tid; d < WDIM_; d += 128)
        emb[row * EMBP_ + d] = (_Float16)wv[(size_t)wid * WDIM_ + d];
    if (tid < EMBP_ - EMB_)
        emb[row * EMBP_ + EMB_ + tid] = (_Float16)0.0f;
}

// ---------------- f32 -> f16 with zero K padding ---------------------------
__global__ void cvt_pad(const float* __restrict__ src, _Float16* __restrict__ dst,
                        int N, int K, int Kp)
{
    const size_t i = (size_t)blockIdx.x * blockDim.x + threadIdx.x;
    const size_t total = (size_t)N * Kp;
    if (i >= total) return;
    const int k = (int)(i % Kp);
    const size_t n = i / Kp;
    dst[i] = (k < K) ? (_Float16)src[n * (size_t)K + k] : (_Float16)0.0f;
}

__global__ void add2(const float* __restrict__ a, const float* __restrict__ b,
                     float* __restrict__ o, int n)
{
    const int i = blockIdx.x * blockDim.x + threadIdx.x;
    if (i < n) o[i] = a[i] + b[i];
}

// ---------------- per-row dot with 200-dim vector --------------------------
__global__ __launch_bounds__(32)
void rowdot(const float* __restrict__ X, const float* __restrict__ w,
            float* __restrict__ out, int D)
{
    __shared__ float r[32];
    const int tid = threadIdx.x;
    const size_t row = blockIdx.x;
    float s = 0.0f;
    for (int d = tid; d < D; d += 32) s += X[row * D + d] * w[d];
    r[tid] = s;
    __syncthreads();
    if (tid == 0) {
        float t = 0.0f;
        for (int i = 0; i < 32; ++i) t += r[i];
        out[row] = t;
    }
}

// ---------------- BiDAF attention row: softmax over Q, U_t, row-max --------
__global__ __launch_bounds__(64)
void attn_row(const float* __restrict__ Hc, const float* __restrict__ Uq,
              const float* __restrict__ cproj, const float* __restrict__ qproj,
              const float* __restrict__ w3, const float* __restrict__ simb,
              float* __restrict__ Ut, float* __restrict__ Smax)
{
    __shared__ float hw[D2_];
    __shared__ float sv[QLEN_];
    const int tid = threadIdx.x;
    const size_t row = blockIdx.x;          // b*CLEN + i
    const int b = (int)(row / CLEN_);

    for (int d = tid; d < D2_; d += 64) hw[d] = Hc[row * D2_ + d] * w3[d];
    __syncthreads();
    for (int j = tid; j < QLEN_; j += 64) {
        const float* uq = Uq + ((size_t)b * QLEN_ + j) * D2_;
        float s = cproj[row] + qproj[b * QLEN_ + j] + simb[0];
        for (int d = 0; d < D2_; ++d) s += hw[d] * uq[d];
        sv[j] = s;
    }
    __syncthreads();
    if (tid == 0) {
        float m = sv[0];
        for (int j = 1; j < QLEN_; ++j) m = fmaxf(m, sv[j]);
        float ssum = 0.0f;
        for (int j = 0; j < QLEN_; ++j) { const float ev = expf(sv[j] - m); sv[j] = ev; ssum += ev; }
        const float inv = 1.0f / ssum;
        for (int j = 0; j < QLEN_; ++j) sv[j] *= inv;
        Smax[row] = m;
    }
    __syncthreads();
    for (int d = tid; d < D2_; d += 64) {
        float s = 0.0f;
        for (int j = 0; j < QLEN_; ++j) s += sv[j] * Uq[((size_t)b * QLEN_ + j) * D2_ + d];
        Ut[row * D2_ + d] = s;
    }
}

// ---------------- softmax over context of row-max + h_t --------------------
__global__ __launch_bounds__(128)
void batt_ht(const float* __restrict__ Smax, const float* __restrict__ Hc,
             float* __restrict__ ht)
{
    __shared__ float sm[CLEN_];
    __shared__ float inv_s;
    const int tid = threadIdx.x;
    const int b = blockIdx.x;
    for (int i = tid; i < CLEN_; i += 128) sm[i] = Smax[(size_t)b * CLEN_ + i];
    __syncthreads();
    if (tid == 0) {
        float m = sm[0];
        for (int i = 1; i < CLEN_; ++i) m = fmaxf(m, sm[i]);
        float ssum = 0.0f;
        for (int i = 0; i < CLEN_; ++i) { const float ev = expf(sm[i] - m); sm[i] = ev; ssum += ev; }
        inv_s = 1.0f / ssum;
    }
    __syncthreads();
    for (int d = tid; d < D2_; d += 128) {
        float s = 0.0f;
        for (int i = 0; i < CLEN_; ++i) s += sm[i] * Hc[((size_t)b * CLEN_ + i) * D2_ + d];
        ht[b * D2_ + d] = s * inv_s;
    }
}

// ---------------- G = [Hc, U_t, Hc*U_t, Hc*h_t] written as f16 -------------
__global__ __launch_bounds__(256)
void build_g(const float* __restrict__ Hc, const float* __restrict__ Ut,
             const float* __restrict__ ht, _Float16* __restrict__ Gh)
{
    const int tid = threadIdx.x;
    const size_t row = blockIdx.x;
    const int b = (int)(row / CLEN_);
    _Float16* g = Gh + row * D8_;
    for (int d = tid; d < D2_; d += 256) {
        const float hc = Hc[row * D2_ + d];
        const float ut = Ut[row * D2_ + d];
        g[d]            = (_Float16)hc;
        g[D2_ + d]      = (_Float16)ut;
        g[2 * D2_ + d]  = (_Float16)(hc * ut);
        g[3 * D2_ + d]  = (_Float16)(hc * ht[b * D2_ + d]);
    }
}

// ---------------- p1/p2 = GM @ w + b ---------------------------------------
__global__ __launch_bounds__(256)
void final_p(const _Float16* __restrict__ Gh, const float* __restrict__ M,
             const float* __restrict__ p1w, const float* __restrict__ p1b,
             const float* __restrict__ p2w, const float* __restrict__ p2b,
             float* __restrict__ out)
{
    __shared__ float r1[256], r2[256];
    const int tid = threadIdx.x;
    const size_t row = blockIdx.x;
    float s1 = 0.0f, s2 = 0.0f;
    for (int d = tid; d < 10 * HID_; d += 256) {
        const float v = (d < D8_) ? (float)Gh[row * D8_ + d] : M[row * D2_ + (d - D8_)];
        s1 += v * p1w[d];
        s2 += v * p2w[d];
    }
    r1[tid] = s1; r2[tid] = s2;
    __syncthreads();
    for (int off = 128; off > 0; off >>= 1) {
        if (tid < off) { r1[tid] += r1[tid + off]; r2[tid] += r2[tid + off]; }
        __syncthreads();
    }
    if (tid == 0) {
        out[row]                        = r1[0] + p1b[0];
        out[(size_t)B_ * CLEN_ + row]   = r2[0] + p2b[0];
    }
}

// ===========================================================================
extern "C" void kernel_launch(void* const* d_in, const int* in_sizes, int n_in,
                              void* d_out, int out_size, void* d_ws, size_t ws_size,
                              hipStream_t stream)
{
    (void)in_sizes; (void)n_in; (void)out_size; (void)ws_size;

    const int*   c_word = (const int*)d_in[0];
    const int*   c_char = (const int*)d_in[1];
    const int*   q_word = (const int*)d_in[2];
    const int*   q_char = (const int*)d_in[3];
    const float* wv     = (const float*)d_in[4];
    const float* ctab   = (const float*)d_in[5];
    const float* cnnw   = (const float*)d_in[6];
    const float* cnnb   = (const float*)d_in[7];
    // ctx_lstm = ((Wih,Whh,bih,bhh) fwd, (..) bwd): inputs 8..15
    const float* ctxF_Wih = (const float*)d_in[8];
    const float* ctxF_Whh = (const float*)d_in[9];
    const float* ctxF_bih = (const float*)d_in[10];
    const float* ctxF_bhh = (const float*)d_in[11];
    const float* ctxB_Wih = (const float*)d_in[12];
    const float* ctxB_Whh = (const float*)d_in[13];
    const float* ctxB_bih = (const float*)d_in[14];
    const float* ctxB_bhh = (const float*)d_in[15];
    const float* simw = (const float*)d_in[16];
    const float* simb = (const float*)d_in[17];
    // mod_lstm = ((l0f, l0b), (l1f, l1b)): inputs 18..33
    const float* m0F_Wih = (const float*)d_in[18];
    const float* m0F_Whh = (const float*)d_in[19];
    const float* m0F_bih = (const float*)d_in[20];
    const float* m0F_bhh = (const float*)d_in[21];
    const float* m0B_Wih = (const float*)d_in[22];
    const float* m0B_Whh = (const float*)d_in[23];
    const float* m0B_bih = (const float*)d_in[24];
    const float* m0B_bhh = (const float*)d_in[25];
    const float* m1F_Wih = (const float*)d_in[26];
    const float* m1F_Whh = (const float*)d_in[27];
    const float* m1F_bih = (const float*)d_in[28];
    const float* m1F_bhh = (const float*)d_in[29];
    const float* m1B_Wih = (const float*)d_in[30];
    const float* m1B_Whh = (const float*)d_in[31];
    const float* m1B_bih = (const float*)d_in[32];
    const float* m1B_bhh = (const float*)d_in[33];
    const float* p1w = (const float*)d_in[34];
    const float* p1b = (const float*)d_in[35];
    const float* p2w = (const float*)d_in[36];
    const float* p2b = (const float*)d_in[37];

    const int RC = B_ * CLEN_;   // 12800 context rows
    const int RQ = B_ * QLEN_;   // 1536 query rows

    // ------------- workspace layout (~124 MB) ------------------------------
    char* ws = (char*)d_ws;
    size_t off = 0;
    auto alloc = [&](size_t bytes) -> char* {
        off = (off + 255) & ~(size_t)255;
        char* p = ws + off;
        off += bytes;
        return p;
    };
    _Float16* cembH = (_Float16*)alloc((size_t)RC * EMBP_ * 2);
    _Float16* qembH = (_Float16*)alloc((size_t)RQ * EMBP_ * 2);
    _Float16* wCtxF_ih = (_Float16*)alloc((size_t)G4_ * EMBP_ * 2);
    _Float16* wCtxB_ih = (_Float16*)alloc((size_t)G4_ * EMBP_ * 2);
    _Float16* wCtxF_hh = (_Float16*)alloc((size_t)G4_ * HP_ * 2);
    _Float16* wCtxB_hh = (_Float16*)alloc((size_t)G4_ * HP_ * 2);
    _Float16* wM0F_ih  = (_Float16*)alloc((size_t)G4_ * D8_ * 2);
    _Float16* wM0B_ih  = (_Float16*)alloc((size_t)G4_ * D8_ * 2);
    _Float16* wM0F_hh  = (_Float16*)alloc((size_t)G4_ * HP_ * 2);
    _Float16* wM0B_hh  = (_Float16*)alloc((size_t)G4_ * HP_ * 2);
    _Float16* wM1F_ih  = (_Float16*)alloc((size_t)G4_ * D2P_ * 2);
    _Float16* wM1B_ih  = (_Float16*)alloc((size_t)G4_ * D2P_ * 2);
    _Float16* wM1F_hh  = (_Float16*)alloc((size_t)G4_ * HP_ * 2);
    _Float16* wM1B_hh  = (_Float16*)alloc((size_t)G4_ * HP_ * 2);
    float* bCtxF = (float*)alloc(G4_ * 4);
    float* bCtxB = (float*)alloc(G4_ * 4);
    float* bM0F  = (float*)alloc(G4_ * 4);
    float* bM0B  = (float*)alloc(G4_ * 4);
    float* bM1F  = (float*)alloc(G4_ * 4);
    float* bM1B  = (float*)alloc(G4_ * 4);
    float* Pa = (float*)alloc((size_t)RC * G4_ * 4);
    float* Pb = (float*)alloc((size_t)RC * G4_ * 4);
    float* Hc = (float*)alloc((size_t)RC * D2_ * 4);
    float* Uq = (float*)alloc((size_t)RQ * D2_ * 4);
    float* cproj = (float*)alloc((size_t)RC * 4);
    float* qproj = (float*)alloc((size_t)RQ * 4);
    float* Smax  = (float*)alloc((size_t)RC * 4);
    float* ht    = (float*)alloc((size_t)B_ * D2_ * 4);
    float* Ut    = (float*)alloc((size_t)RC * D2_ * 4);
    _Float16* Gh  = (_Float16*)alloc((size_t)RC * D8_ * 2);
    float* M1    = (float*)alloc((size_t)RC * D2_ * 4);
    _Float16* M1h = (_Float16*)alloc((size_t)RC * D2P_ * 2);
    float* Mm    = (float*)alloc((size_t)RC * D2_ * 4);

    auto cvt = [&](const float* s, _Float16* d, int N, int K, int Kp) {
        const int total = N * Kp;
        cvt_pad<<<(total + 255) / 256, 256, 0, stream>>>(s, d, N, K, Kp);
    };
    auto gemm = [&](const _Float16* A, const _Float16* W, float* C, int M, int N, int Kp) {
        dim3 g(N / 80, M / 16);   // N == 400 for every GEMM in this net
        gemm_wmma_f16<<<g, 32, 0, stream>>>(A, W, C, M, N, Kp);
    };

    // ---- weight / bias prep (deterministic, recomputed each call) ---------
    cvt(ctxF_Wih, wCtxF_ih, G4_, EMB_, EMBP_);
    cvt(ctxB_Wih, wCtxB_ih, G4_, EMB_, EMBP_);
    cvt(ctxF_Whh, wCtxF_hh, G4_, HID_, HP_);
    cvt(ctxB_Whh, wCtxB_hh, G4_, HID_, HP_);
    cvt(m0F_Wih,  wM0F_ih,  G4_, D8_,  D8_);
    cvt(m0B_Wih,  wM0B_ih,  G4_, D8_,  D8_);
    cvt(m0F_Whh,  wM0F_hh,  G4_, HID_, HP_);
    cvt(m0B_Whh,  wM0B_hh,  G4_, HID_, HP_);
    cvt(m1F_Wih,  wM1F_ih,  G4_, D2_,  D2P_);
    cvt(m1B_Wih,  wM1B_ih,  G4_, D2_,  D2P_);
    cvt(m1F_Whh,  wM1F_hh,  G4_, HID_, HP_);
    cvt(m1B_Whh,  wM1B_hh,  G4_, HID_, HP_);
    add2<<<2, 256, 0, stream>>>(ctxF_bih, ctxF_bhh, bCtxF, G4_);
    add2<<<2, 256, 0, stream>>>(ctxB_bih, ctxB_bhh, bCtxB, G4_);
    add2<<<2, 256, 0, stream>>>(m0F_bih,  m0F_bhh,  bM0F,  G4_);
    add2<<<2, 256, 0, stream>>>(m0B_bih,  m0B_bhh,  bM0B,  G4_);
    add2<<<2, 256, 0, stream>>>(m1F_bih,  m1F_bhh,  bM1F,  G4_);
    add2<<<2, 256, 0, stream>>>(m1B_bih,  m1B_bhh,  bM1B,  G4_);

    // ---- embeddings (word gather + char CNN), f16 padded ------------------
    embed_kernel<<<RC, 128, 0, stream>>>(c_word, c_char, wv, ctab, cnnw, cnnb, cembH);
    embed_kernel<<<RQ, 128, 0, stream>>>(q_word, q_char, wv, ctab, cnnw, cnnb, qembH);

    // ---- context BiLSTM ---------------------------------------------------
    gemm(cembH, wCtxF_ih, Pa, RC, G4_, EMBP_);
    gemm(cembH, wCtxB_ih, Pb, RC, G4_, EMBP_);
    lstm_birec<<<dim3(2, 2), 256, 0, stream>>>(Pa, Pb, wCtxF_hh, wCtxB_hh, bCtxF, bCtxB, Hc, CLEN_);
    gemm(qembH, wCtxF_ih, Pa, RQ, G4_, EMBP_);
    gemm(qembH, wCtxB_ih, Pb, RQ, G4_, EMBP_);
    lstm_birec<<<dim3(2, 2), 256, 0, stream>>>(Pa, Pb, wCtxF_hh, wCtxB_hh, bCtxF, bCtxB, Uq, QLEN_);

    // ---- BiDAF attention --------------------------------------------------
    rowdot<<<RC, 32, 0, stream>>>(Hc, simw,       cproj, D2_);  // w1
    rowdot<<<RQ, 32, 0, stream>>>(Uq, simw + D2_, qproj, D2_);  // w2
    attn_row<<<RC, 64, 0, stream>>>(Hc, Uq, cproj, qproj, simw + 2 * D2_, simb, Ut, Smax);
    batt_ht<<<B_, 128, 0, stream>>>(Smax, Hc, ht);
    build_g<<<RC, 256, 0, stream>>>(Hc, Ut, ht, Gh);

    // ---- modeling BiLSTM stack -------------------------------------------
    gemm(Gh, wM0F_ih, Pa, RC, G4_, D8_);
    gemm(Gh, wM0B_ih, Pb, RC, G4_, D8_);
    lstm_birec<<<dim3(2, 2), 256, 0, stream>>>(Pa, Pb, wM0F_hh, wM0B_hh, bM0F, bM0B, M1, CLEN_);
    cvt(M1, M1h, RC, D2_, D2P_);
    gemm(M1h, wM1F_ih, Pa, RC, G4_, D2P_);
    gemm(M1h, wM1B_ih, Pb, RC, G4_, D2P_);
    lstm_birec<<<dim3(2, 2), 256, 0, stream>>>(Pa, Pb, wM1F_hh, wM1B_hh, bM1F, bM1B, Mm, CLEN_);

    // ---- output projections ----------------------------------------------
    final_p<<<RC, 256, 0, stream>>>(Gh, Mm, p1w, p1b, p2w, p2b, (float*)d_out);
}
